// AttentionBase_82927228551364
// MI455X (gfx1250) — matrix operand order repaired and checked by reading, not compiled, not run on hardware
//
#include <hip/hip_runtime.h>
#include <hip/hip_bf16.h>

#define BATCH 4
#define SEQ   2048
#define HEADS 8
#define DHEAD 64
#define MID   512            // HEADS*DHEAD == FEATURES
#define NK    (SEQ + 1)      // 2049 keys incl. null token
#define NKP   2112           // padded key count: 33*64, covers jcol<=2079, 32B-aligned rows

typedef __attribute__((ext_vector_type(16))) __bf16 v16bf;
typedef __attribute__((ext_vector_type(8)))  __bf16 v8bf;
typedef __attribute__((ext_vector_type(8)))  float  v8f;

union V16U { v16bf v; v8bf h[2]; };

// ---------- helpers ----------
__device__ __forceinline__ unsigned short f2bfu(float f) {
  unsigned u = __builtin_bit_cast(unsigned, f);
  u += 0x7fffu + ((u >> 16) & 1u);          // round-to-nearest-even
  return (unsigned short)(u >> 16);
}
__device__ __forceinline__ __bf16 f2bf(float f) {
  unsigned short h = f2bfu(f);
  return __builtin_bit_cast(__bf16, h);
}
// 16-bit A-fragment K index for element e (contiguous, matches B loads):
// lanes 0-15: K = {0..15}; lanes 16-31: K = {16..31}
__device__ __forceinline__ int kmapAB(int lane, int e) {
  return e + ((lane >= 16) ? 16 : 0);
}
__device__ __forceinline__ float rmax16(float v) {
#pragma unroll
  for (int m = 1; m < 16; m <<= 1) v = fmaxf(v, __shfl_xor(v, m, 32));
  return v;
}
__device__ __forceinline__ float rsum16(float v) {
#pragma unroll
  for (int m = 1; m < 16; m <<= 1) v += __shfl_xor(v, m, 32);
  return v;
}
__device__ __forceinline__ v8f wmma_bf16(v16bf a, v16bf b, v8f c) {
  return __builtin_amdgcn_wmma_f32_16x16x32_bf16(false, a, false, b, (short)0, c,
                                                 false, false);
}

// =====================================================================
// Pre-pass 1: k,v (f32, [b,n,h*d]) -> bf16 K[b,h,j,d] and V^T[b,h,d,j],
// null token at j=0, zero padding for j in (SEQ, NKP).
// grid (NKP/64, HEADS, BATCH), block 256.
// =====================================================================
__global__ __launch_bounds__(256) void conv_kv_kernel(
    const float* __restrict__ k, const float* __restrict__ v,
    const float* __restrict__ tokens, unsigned short* __restrict__ kbf,
    unsigned short* __restrict__ vbfT) {
  __shared__ unsigned short vt[64][65];     // +1 pad: transpose bank spread

  const int tid = threadIdx.x;
  const int bb  = blockIdx.z, hh = blockIdx.y;
  const int bh  = bb * HEADS + hh;
  const int j0  = blockIdx.x * 64;

  // phase 1: coalesced f32 reads, convert, write K rows + stage V tile
  const int jr = tid >> 2;                  // 0..63  (tile row = key)
  const int db = (tid & 3) * 16;            // 16 consecutive head dims
  const int jcol = j0 + jr;
  const float* src = (jcol > 0 && jcol <= SEQ)
        ? k + ((size_t)bb * SEQ + (jcol - 1)) * MID + hh * DHEAD + db
        : nullptr;
  const float* srcv = (jcol > 0 && jcol <= SEQ)
        ? v + ((size_t)bb * SEQ + (jcol - 1)) * MID + hh * DHEAD + db
        : nullptr;
#pragma unroll
  for (int e = 0; e < 16; ++e) {
    const int d = db + e;
    float kv, vv;
    if (jcol == 0)           { kv = tokens[d]; vv = tokens[DHEAD + d]; }
    else if (jcol <= SEQ)    { kv = src[e];    vv = srcv[e]; }
    else                     { kv = 0.f;       vv = 0.f; }
    kbf[((size_t)bh * NKP + jcol) * DHEAD + d] = f2bfu(kv);
    vt[jr][d] = f2bfu(vv);
  }
  __syncthreads();

  // phase 2: transposed, coalesced bf16 writes of V^T
  const int dr  = tid >> 2;                 // head dim
  const int jb2 = (tid & 3) * 16;           // 16 consecutive keys
  unsigned short* dst = vbfT + ((size_t)bh * DHEAD + dr) * NKP + j0 + jb2;
#pragma unroll
  for (int e = 0; e < 16; ++e) dst[e] = vt[jb2 + e][dr];
}

// =====================================================================
// Pre-pass 2: w_out f32 -> bf16 (same layout [f, c]).
// grid 256, block 256, 4 elems/thread.
// =====================================================================
__global__ __launch_bounds__(256) void conv_w_kernel(
    const float* __restrict__ w_out, unsigned short* __restrict__ wbf) {
  const int idx = (blockIdx.x * 256 + threadIdx.x) * 4;
#pragma unroll
  for (int e = 0; e < 4; ++e) wbf[idx + e] = f2bfu(w_out[idx + e]);
}

// =====================================================================
// Kernel A: flash attention.  One wave = 16 query rows of one (b,h).
// grid (16, HEADS, BATCH), block 256 (8 waves).
// =====================================================================
__global__ __launch_bounds__(256) void attn_fa_kernel(
    const float* __restrict__ q, const unsigned short* __restrict__ kbf,
    const unsigned short* __restrict__ vbfT,
    const unsigned char* __restrict__ mask, const float* __restrict__ bias,
    unsigned short* __restrict__ o_ws) {
  __shared__ unsigned short pLds[8][16 * 32];   // per-wave P staging tile

  const int lane = threadIdx.x & 31;
  const int wv   = threadIdx.x >> 5;
  const int bb   = blockIdx.z;
  const int hh   = blockIdx.y;
  const int i0   = blockIdx.x * 128 + wv * 16;  // first query row of this wave
  const int l15  = lane & 15;
  const int hi   = lane >> 4;                   // 0 or 1
  const int bh   = bb * HEADS + hh;

  // ---- Q tile as two 16x32 bf16 A-fragments (scale folded in; one-time) ----
  const float* qrow = q + ((size_t)(bb * SEQ + i0 + l15)) * MID + hh * DHEAD;
  v16bf qa[2];
#pragma unroll
  for (int kc = 0; kc < 2; ++kc) {
#pragma unroll
    for (int e = 0; e < 16; ++e)
      qa[kc][e] = f2bf(qrow[kc * 32 + kmapAB(lane, e)] * 0.125f);
  }

  v8f oacc[4];
#pragma unroll
  for (int t = 0; t < 4; ++t) oacc[t] = {};
  float mrow[8], lrow[8];
#pragma unroll
  for (int r = 0; r < 8; ++r) { mrow[r] = -3.0e38f; lrow[r] = 0.f; }

  const float* bias_base = bias + ((size_t)(bh * SEQ)) * NK;
  const unsigned char* mk = mask + (size_t)bb * SEQ;
  const unsigned short* kb_bh = kbf  + (size_t)bh * NKP * DHEAD;
  const unsigned short* vt_bh = vbfT + (size_t)bh * DHEAD * NKP;
  unsigned short* pl = &pLds[wv][0];

  for (int jt = 0; jt < (NK + 31) / 32; ++jt) {
    const int jb = jt * 32;
    __builtin_prefetch(bias_base + (size_t)(i0 + l15) * NK + jb + 32, 0, 1);

    // ---- S = Q K^T + bias, masked (fully branch-free) ----
    v8f s[2];
#pragma unroll
    for (int nj = 0; nj < 2; ++nj) {
      const int jcol = jb + nj * 16 + l15;          // this lane's key column
      const unsigned short* kp = kb_bh + (size_t)jcol * DHEAD + hi * 16;
      v8f sa = {};
#pragma unroll
      for (int kc = 0; kc < 2; ++kc) {              // aligned 32B vector loads
        const v16bf kfrag = *(const v16bf*)(const void*)(kp + kc * 32);
        sa = wmma_bf16(qa[kc], kfrag, sa);
      }
      // unconditional clamped mask-byte load; bitwise (non-short-circuit) predicate
      int midx = jcol - 1;
      midx = midx < 0 ? 0 : midx;
      midx = midx > (SEQ - 1) ? (SEQ - 1) : midx;
      const unsigned char mb = mk[midx];
      const int valid = ((int)(jcol <= SEQ)) & (((int)(jcol == 0)) | ((int)(mb != 0)));
      const int jc = jcol > SEQ ? SEQ : jcol;       // clamp for bias read only
#pragma unroll
      for (int r = 0; r < 8; ++r) {
        const int ig = i0 + r + hi * 8;             // C/D layout row
        const float bv = bias_base[(size_t)ig * NK + jc];
        const float sv = sa[r] + bv;                // computed unconditionally
        sa[r] = valid ? sv : -3.0e38f;              // -> v_cndmask
      }
      s[nj] = sa;
    }

    // ---- online softmax (per-row m/l), rescale O, stage P in LDS ----
#pragma unroll
    for (int r = 0; r < 8; ++r) {
      const float s0 = s[0][r], s1 = s[1][r];
      const float mn = fmaxf(mrow[r], rmax16(fmaxf(s0, s1)));
      const float p0 = __expf(s0 - mn);
      const float p1 = __expf(s1 - mn);
      const float rs = rsum16(p0 + p1);
      const float al = __expf(mrow[r] - mn);
      lrow[r] = lrow[r] * al + rs;
      mrow[r] = mn;
#pragma unroll
      for (int t = 0; t < 4; ++t) oacc[t][r] *= al;
      const int M = r + hi * 8;
      pl[M * 32 + l15]      = f2bfu(p0);
      pl[M * 32 + 16 + l15] = f2bfu(p1);
    }
    __builtin_amdgcn_wave_barrier();   // LDS in-order per wave; fence scheduling

    // ---- P A-fragment: two contiguous 16B runs from LDS ----
    V16U pu;
    pu.h[0] = *(const v8bf*)(const void*)(pl + l15 * 32 + hi * 16);
    pu.h[1] = *(const v8bf*)(const void*)(pl + l15 * 32 + hi * 16 + 8);
    const v16bf pa = pu.v;
    __builtin_amdgcn_wave_barrier();

    // ---- O += P * V  (V^T rows -> aligned 32B vector loads) ----
#pragma unroll
    for (int t = 0; t < 4; ++t) {
      const int dd = t * 16 + l15;
      const v16bf vfrag =
          *(const v16bf*)(const void*)(vt_bh + (size_t)dd * NKP + jb + hi * 16);
      oacc[t] = wmma_bf16(pa, vfrag, oacc[t]);
    }
  }

  // ---- normalize and spill O (bf16) to workspace ----
#pragma unroll
  for (int t = 0; t < 4; ++t) {
#pragma unroll
    for (int r = 0; r < 8; ++r) {
      const int ig = i0 + r + hi * 8;
      const int dd = t * 16 + l15;
      const float val = oacc[t][r] / lrow[r];
      o_ws[((size_t)(bb * SEQ + ig)) * MID + hh * DHEAD + dd] = f2bfu(val);
    }
  }
}

// =====================================================================
// Kernel B: out = LN(O @ w_out^T) * gamma.  16 rows per block, 8 waves,
// each wave computes 64 output features over the full K=512.
// =====================================================================
__global__ __launch_bounds__(256) void proj_ln_kernel(
    const unsigned short* __restrict__ o_ws,
    const unsigned short* __restrict__ wbf, const float* __restrict__ gamma,
    float* __restrict__ out) {
  __shared__ unsigned short oLds[16 * MID];     // 16 KB bf16 tile
  __shared__ float rsumS[16], rsqS[16];

  const int tid  = threadIdx.x;
  const int lane = tid & 31;
  const int wv   = tid >> 5;
  const int l15  = lane & 15;
  const int hi   = lane >> 4;
  const int row0 = blockIdx.x * 16;

  {   // cooperative 16x512 bf16 tile load (4 x uint4 per thread)
    const uint4* src = (const uint4*)(o_ws + (size_t)row0 * MID);
    uint4* dst = (uint4*)oLds;
#pragma unroll
    for (int i = 0; i < 4; ++i) dst[tid + i * 256] = src[tid + i * 256];
  }
  if (tid < 16) { rsumS[tid] = 0.f; rsqS[tid] = 0.f; }
  __syncthreads();

  const int cb = wv * 64;                       // wave's feature base
  v8f acc[4];
#pragma unroll
  for (int t = 0; t < 4; ++t) acc[t] = {};

  for (int kc = 0; kc < 16; ++kc) {
    V16U au;                                    // A frag = two 16B LDS runs
    const unsigned short* ap = oLds + l15 * MID + kc * 32 + hi * 16;
    au.h[0] = *(const v8bf*)(const void*)(ap);
    au.h[1] = *(const v8bf*)(const void*)(ap + 8);
    const v16bf aa = au.v;
    const int crow = kc * 32 + hi * 16;
#pragma unroll
    for (int t = 0; t < 4; ++t) {
      const int f = cb + t * 16 + l15;
      const v16bf bfrag =                       // B(c,f) = w_out[f,c], bf16
          *(const v16bf*)(const void*)(wbf + (size_t)f * MID + crow);
      acc[t] = wmma_bf16(aa, bfrag, acc[t]);
    }
  }

  // ---- LayerNorm row statistics (shfl + shared ds_add_f32) ----
#pragma unroll
  for (int r = 0; r < 8; ++r) {
    float sv = 0.f, sq = 0.f;
#pragma unroll
    for (int t = 0; t < 4; ++t) { const float x = acc[t][r]; sv += x; sq += x * x; }
    sv = rsum16(sv); sq = rsum16(sq);
    if (l15 == 0) {
      atomicAdd(&rsumS[r + hi * 8], sv);
      atomicAdd(&rsqS[r + hi * 8], sq);
    }
  }
  __syncthreads();

  float meanr[8], invr[8];
#pragma unroll
  for (int r = 0; r < 8; ++r) {
    const int M = r + hi * 8;
    const float mu  = rsumS[M] * (1.0f / MID);
    const float var = rsqS[M] * (1.0f / MID) - mu * mu;
    meanr[r] = mu;
    invr[r]  = rsqrtf(var + 1e-5f);
  }
#pragma unroll
  for (int t = 0; t < 4; ++t) {
    const int f = cb + t * 16 + l15;
    const float g = gamma[f];
#pragma unroll
    for (int r = 0; r < 8; ++r) {
      const int M = r + hi * 8;
      out[((size_t)(row0 + M)) * MID + f] = (acc[t][r] - meanr[r]) * invr[r] * g;
    }
  }
}

extern "C" void kernel_launch(void* const* d_in, const int* in_sizes, int n_in,
                              void* d_out, int out_size, void* d_ws, size_t ws_size,
                              hipStream_t stream) {
  (void)in_sizes; (void)n_in; (void)out_size; (void)ws_size;
  const float*         q      = (const float*)d_in[0];
  const float*         k      = (const float*)d_in[1];
  const float*         v      = (const float*)d_in[2];
  const unsigned char* mask   = (const unsigned char*)d_in[3];  // jax bool -> 1B
  const float*         bias   = (const float*)d_in[4];
  const float*         tokens = (const float*)d_in[5];
  const float*         w_out  = (const float*)d_in[6];
  const float*         gamma  = (const float*)d_in[7];

  // workspace layout (bytes): O | Kbf | VbfT | Wbf  (~25 MB total)
  char* ws = (char*)d_ws;
  const size_t O_bytes = (size_t)BATCH * SEQ * MID * 2;                 // 8 MB
  const size_t K_bytes = (size_t)BATCH * HEADS * NKP * DHEAD * 2;       // 8.25 MB
  unsigned short* o_ws = (unsigned short*)(ws);
  unsigned short* kbf  = (unsigned short*)(ws + O_bytes);
  unsigned short* vbfT = (unsigned short*)(ws + O_bytes + K_bytes);
  unsigned short* wbf  = (unsigned short*)(ws + O_bytes + 2 * K_bytes);

  conv_kv_kernel<<<dim3(NKP / 64, HEADS, BATCH), 256, 0, stream>>>(
      k, v, tokens, kbf, vbfT);
  conv_w_kernel<<<dim3(256), 256, 0, stream>>>(w_out, wbf);
  attn_fa_kernel<<<dim3(16, HEADS, BATCH), 256, 0, stream>>>(
      q, kbf, vbfT, mask, bias, o_ws);
  proj_ln_kernel<<<dim3((BATCH * SEQ) / 16), 256, 0, stream>>>(
      o_ws, wbf, gamma, (float*)d_out);
}
